// NKDGNN_38654705664211
// MI455X (gfx1250) — matrix operand order repaired and checked by reading, not compiled, not run on hardware
//
#include <hip/hip_runtime.h>
#include <math.h>

typedef __bf16 bf16_t;
typedef __attribute__((ext_vector_type(16))) __bf16 v16bf;
typedef __attribute__((ext_vector_type(8)))  __bf16 v8bf;
typedef __attribute__((ext_vector_type(8)))  float  v8f;

// ---------------- helpers ----------------
static __device__ __forceinline__ unsigned f2key(float f) {
  unsigned b = __float_as_uint(f);
  return (b & 0x80000000u) ? ~b : (b | 0x80000000u);
}
static __device__ __forceinline__ float key2f(unsigned k) {
  unsigned b = (k & 0x80000000u) ? (k & 0x7fffffffu) : ~k;
  return __uint_as_float(b);
}

// ---------------- utility kernels ----------------
__global__ void fill_u32_kernel(unsigned* __restrict__ p, unsigned v, size_t n) {
  size_t i = (size_t)blockIdx.x * blockDim.x + threadIdx.x;
  size_t st = (size_t)gridDim.x * blockDim.x;
  for (; i < n; i += st) p[i] = v;
}

__global__ void f32_to_bf16_kernel(const float* __restrict__ s, bf16_t* __restrict__ d, size_t n) {
  size_t i = (size_t)blockIdx.x * blockDim.x + threadIdx.x;
  size_t st = (size_t)gridDim.x * blockDim.x;
  for (; i < n; i += st) d[i] = (bf16_t)s[i];
}

// src is [R,C] row-major; dst becomes [C,R] row-major (bf16)
__global__ void transpose_f32_to_bf16_kernel(const float* __restrict__ s, bf16_t* __restrict__ d,
                                             int R, int C) {
  int t = blockIdx.x * blockDim.x + threadIdx.x;
  if (t >= R * C) return;
  int r = t / C, c = t - r * C;
  d[(size_t)c * R + r] = (bf16_t)s[t];
}

// ---------------- WMMA GEMM: C[M,N] = A[M,K] x Bt[N,K]^T (+bias)(+relu) ----------------
// One wave per 16x16 C tile. A,Bt bf16 row-major. M%16==0, N%16==0, K%32==0.
__global__ __launch_bounds__(256) void wmma_gemm_bf16_kernel(
    const bf16_t* __restrict__ A, const bf16_t* __restrict__ Bt,
    float* __restrict__ C, const float* __restrict__ bias,
    int M, int N, int K, int relu) {
  int tilesN = N >> 4;
  int wave = (int)(((size_t)blockIdx.x * blockDim.x + threadIdx.x) >> 5);
  int total = (M >> 4) * tilesN;
  if (wave >= total) return;
  int tm = wave / tilesN, tn = wave - tm * tilesN;
  int lane = threadIdx.x & 31;
  int g = lane >> 4;        // half-wave group
  int r16 = lane & 15;
  const bf16_t* arow = A + (size_t)(tm * 16 + r16) * K;   // A-frag: M = lane&15
  const bf16_t* brow = Bt + (size_t)(tn * 16 + r16) * K;  // B-frag: N = lane&15
  v8f acc = {};
  for (int k0 = 0; k0 < K; k0 += 32) {
    // A 16x32 layout (matches B): lane-group g holds K = k0+16g .. k0+16g+15 (contiguous in A row)
    v16bf af = *(const v16bf*)(arow + k0 + 16 * g);
    // B 32x16 layout (matches A): lane-group g holds K = k0+16g .. k0+16g+15 (contiguous in Bt row)
    v16bf bfrag = *(const v16bf*)(brow + k0 + 16 * g);
    if (k0 + 32 < K) {
      __builtin_prefetch(arow + k0 + 32, 0, 1);
      __builtin_prefetch(brow + k0 + 32, 0, 1);
    }
    acc = __builtin_amdgcn_wmma_f32_16x16x32_bf16(false, af, false, bfrag,
                                                  (short)0, acc, false, false);
  }
  int col = tn * 16 + r16;
  float bb = bias ? bias[col] : 0.0f;
#pragma unroll
  for (int r = 0; r < 8; ++r) {
    int row = tm * 16 + r + 8 * g;   // C layout: VGPR r -> M=r (lanes 0-15) / M=r+8 (lanes 16-31)
    float v = acc[r] + bb;
    if (relu) v = fmaxf(v, 0.0f);
    C[(size_t)row * N + col] = v;
  }
}

// ---------------- GAT edge phase ----------------
// alpha_s[n,h] = sum_c hlin[n,h*Cc+c]*avs[h*Cc+c]; same for alpha_d
__global__ void alpha_kernel(const float* __restrict__ hlin, const float* __restrict__ avs,
                             const float* __restrict__ avd, float* __restrict__ as_,
                             float* __restrict__ ad_, int Nn, int H, int Cc) {
  int t = blockIdx.x * blockDim.x + threadIdx.x;
  if (t >= Nn * H) return;
  int n = t / H, h = t - n * H;
  const float* row = hlin + (size_t)n * H * Cc + (size_t)h * Cc;
  float ss = 0.f, sd = 0.f;
  for (int c = 0; c < Cc; ++c) {
    float v = row[c];
    ss += v * avs[h * Cc + c];
    sd += v * avd[h * Cc + c];
  }
  as_[t] = ss;
  ad_[t] = sd;
}

static __device__ __forceinline__ void edge_nodes(const long long* ei, int E, int e, int& s, int& d) {
  if (e < E) { s = (int)ei[e]; d = (int)ei[E + e]; }
  else       { s = e - E; d = e - E; }           // self-loop
}

__global__ void edge_max_kernel(const long long* __restrict__ ei, int E, int Nn,
                                const float* __restrict__ as_, const float* __restrict__ ad_,
                                unsigned* __restrict__ mkey, int H) {
  int t = blockIdx.x * blockDim.x + threadIdx.x;
  int total = (E + Nn) * H;
  if (t >= total) return;
  int e = t / H, h = t - e * H;
  int s, d; edge_nodes(ei, E, e, s, d);
  float ev = as_[s * H + h] + ad_[d * H + h];
  ev = (ev > 0.f) ? ev : 0.2f * ev;              // leaky_relu 0.2
  atomicMax(&mkey[d * H + h], f2key(ev));
}

__global__ void edge_expsum_kernel(const long long* __restrict__ ei, int E, int Nn,
                                   const float* __restrict__ as_, const float* __restrict__ ad_,
                                   const unsigned* __restrict__ mkey, float* __restrict__ ssum, int H) {
  int t = blockIdx.x * blockDim.x + threadIdx.x;
  int total = (E + Nn) * H;
  if (t >= total) return;
  int e = t / H, h = t - e * H;
  int s, d; edge_nodes(ei, E, e, s, d);
  float ev = as_[s * H + h] + ad_[d * H + h];
  ev = (ev > 0.f) ? ev : 0.2f * ev;
  float m = key2f(mkey[d * H + h]);
  atomicAdd(&ssum[d * H + h], expf(ev - m));
}

// one wave per edge; lanes stripe the H*Cc channels; atomic adds land in L2 (tables fit 192MB L2)
__global__ __launch_bounds__(256) void edge_agg_kernel(
    const long long* __restrict__ ei, int E, int Nn,
    const float* __restrict__ as_, const float* __restrict__ ad_,
    const unsigned* __restrict__ mkey, const float* __restrict__ ssum,
    const float* __restrict__ hlin, float* __restrict__ hout, int H, int Cc) {
  int wave = (int)(((size_t)blockIdx.x * blockDim.x + threadIdx.x) >> 5);
  int lane = threadIdx.x & 31;
  int total = E + Nn;
  if (wave >= total) return;
  int s, d; edge_nodes(ei, E, wave, s, d);
  float alphas[4];
  for (int h = 0; h < H; ++h) {
    float ev = as_[s * H + h] + ad_[d * H + h];
    ev = (ev > 0.f) ? ev : 0.2f * ev;
    float m = key2f(mkey[d * H + h]);
    alphas[h] = expf(ev - m) / (ssum[d * H + h] + 1e-16f);
  }
  int Ct = H * Cc;
  const float* hs = hlin + (size_t)s * Ct;
  float* hd = hout + (size_t)d * Ct;
  for (int idx = lane; idx < Ct; idx += 32)
    atomicAdd(&hd[idx], alphas[idx / Cc] * hs[idx]);
}

__global__ void bias_relu_kernel(float* __restrict__ h, const float* __restrict__ b,
                                 size_t total, int D) {
  size_t i = (size_t)blockIdx.x * blockDim.x + threadIdx.x;
  size_t st = (size_t)gridDim.x * blockDim.x;
  for (; i < total; i += st) h[i] = fmaxf(h[i] + b[i % (size_t)D], 0.f);
}

// ---------------- pooling / fusion ----------------
__global__ void pool_sum_kernel(const float* __restrict__ h, const long long* __restrict__ batch,
                                float* __restrict__ psum, float* __restrict__ pcnt, int Nn, int D) {
  int t = blockIdx.x * blockDim.x + threadIdx.x;
  if (t >= Nn * D) return;
  int n = t / D, j = t - n * D;
  int g = (int)batch[n];
  atomicAdd(&psum[g * D + j], h[(size_t)n * D + j]);
  if (j == 0) atomicAdd(&pcnt[g], 1.0f);
}

__global__ void pool_div_kernel(const float* __restrict__ psum, const float* __restrict__ pcnt,
                                float* __restrict__ gf, int G, int D) {
  int t = blockIdx.x * blockDim.x + threadIdx.x;
  if (t >= G * D) return;
  int g = t / D;
  gf[t] = psum[t] / fmaxf(pcnt[g], 1.0f);
}

__global__ void concat_bf16_kernel(const float* __restrict__ a, const float* __restrict__ b,
                                   bf16_t* __restrict__ d, int G, int D) {
  int t = blockIdx.x * blockDim.x + threadIdx.x;
  if (t >= G * 2 * D) return;
  int g = t / (2 * D), j = t - g * (2 * D);
  float v = (j < D) ? a[g * D + j] : b[g * D + (j - D)];
  d[t] = (bf16_t)v;
}

// ---------------- host ----------------
static inline unsigned blk_exact(long long n, int t) { return (unsigned)((n + t - 1) / t); }
static inline unsigned blk_gs(size_t n, int t) {
  size_t b = (n + (size_t)t - 1) / (size_t)t;
  if (b > 262144) b = 262144;
  if (b == 0) b = 1;
  return (unsigned)b;
}

extern "C" void kernel_launch(void* const* d_in, const int* in_sizes, int n_in,
                              void* d_out, int out_size, void* d_ws, size_t ws_size,
                              hipStream_t stream) {
  (void)in_sizes; (void)n_in; (void)out_size; (void)ws_size;
  const int Nn = 50000, E = 800000, G = 64, VISD = 2048, GD = 256, HID = 64;
  const int H1 = 4, C1 = 64, Ct1 = H1 * C1;      // 256
  const int H2 = 1, C2 = 64, Ct2 = H2 * C2;      // 64
  const int Etot = E + Nn;

  const float*     visual = (const float*)d_in[0];
  const float*     x      = (const float*)d_in[1];
  const long long* ei     = (const long long*)d_in[2];
  const long long* batch  = (const long long*)d_in[3];
  const float* w1   = (const float*)d_in[4];
  const float* as1v = (const float*)d_in[5];
  const float* ad1v = (const float*)d_in[6];
  const float* b1   = (const float*)d_in[7];
  const float* w2   = (const float*)d_in[8];
  const float* as2v = (const float*)d_in[9];
  const float* ad2v = (const float*)d_in[10];
  const float* b2   = (const float*)d_in[11];
  const float* wvp  = (const float*)d_in[12];
  const float* bvp  = (const float*)d_in[13];
  // d_in[14..17] = wq,bq,wk,bk : softmax over S=1 makes them dead code
  const float* wv   = (const float*)d_in[18];
  const float* bv   = (const float*)d_in[19];
  const float* wo   = (const float*)d_in[20];
  const float* bo   = (const float*)d_in[21];
  const float* wf   = (const float*)d_in[22];
  const float* bff  = (const float*)d_in[23];
  float* out = (float*)d_out;

  // workspace layout (256B aligned, with reuse between layers)
  char* ws = (char*)d_ws;
  size_t off = 0;
  auto take = [&](size_t bytes) -> size_t {
    size_t r = off; off += (bytes + 255) & ~(size_t)255; return r;
  };
  bf16_t*   Abf  = (bf16_t*)(ws + take((size_t)Nn * Ct1 * 2));        // x_bf16, then relu(h1)_bf16
  bf16_t*   Wbf  = (bf16_t*)(ws + take((size_t)G * VISD * 2));        // largest weight (wvp)
  float*    Hlin = (float*)(ws + take((size_t)Nn * Ct1 * 4));         // h1_lin, then h2_lin
  float*    Hout = (float*)(ws + take((size_t)Nn * Ct1 * 4));         // h1_out, then h2_out
  float*    As   = (float*)(ws + take((size_t)Nn * H1 * 4));
  float*    Ad   = (float*)(ws + take((size_t)Nn * H1 * 4));
  unsigned* Mk   = (unsigned*)(ws + take((size_t)Nn * H1 * 4));
  float*    Ss   = (float*)(ws + take((size_t)Nn * H1 * 4));
  float*    Ps   = (float*)(ws + take((size_t)G * HID * 4));
  float*    Pc   = (float*)(ws + take((size_t)G * 4));
  float*    Gf   = (float*)(ws + take((size_t)G * HID * 4));
  float*    Vis  = (float*)(ws + take((size_t)G * HID * 4));
  float*    Vt   = (float*)(ws + take((size_t)G * HID * 4));
  float*    At   = (float*)(ws + take((size_t)G * HID * 4));
  bf16_t*   Gfb  = (bf16_t*)(ws + take((size_t)G * HID * 2));
  bf16_t*   Vtb  = (bf16_t*)(ws + take((size_t)G * HID * 2));
  bf16_t*   Cmb  = (bf16_t*)(ws + take((size_t)G * 2 * HID * 2));
  bf16_t*   Visb = (bf16_t*)(ws + take((size_t)G * VISD * 2));

  auto gemm = [&](const bf16_t* A, const bf16_t* Bt, float* C, const float* bias,
                  int M, int N, int K, int relu) {
    int tiles = (M / 16) * (N / 16);
    wmma_gemm_bf16_kernel<<<blk_exact(tiles, 8), 256, 0, stream>>>(A, Bt, C, bias, M, N, K, relu);
  };

  // ===== GAT layer 1 =====
  f32_to_bf16_kernel<<<blk_gs((size_t)Nn * GD, 256), 256, 0, stream>>>(x, Abf, (size_t)Nn * GD);
  transpose_f32_to_bf16_kernel<<<blk_exact(GD * Ct1, 256), 256, 0, stream>>>(w1, Wbf, GD, Ct1);
  gemm(Abf, Wbf, Hlin, nullptr, Nn, Ct1, GD, 0);                       // h1_lin = x @ w1
  alpha_kernel<<<blk_exact(Nn * H1, 256), 256, 0, stream>>>(Hlin, as1v, ad1v, As, Ad, Nn, H1, C1);
  fill_u32_kernel<<<blk_gs((size_t)Nn * H1, 256), 256, 0, stream>>>(Mk, 0u, (size_t)Nn * H1);
  fill_u32_kernel<<<blk_gs((size_t)Nn * H1, 256), 256, 0, stream>>>((unsigned*)Ss, 0u, (size_t)Nn * H1);
  fill_u32_kernel<<<blk_gs((size_t)Nn * Ct1, 256), 256, 0, stream>>>((unsigned*)Hout, 0u, (size_t)Nn * Ct1);
  edge_max_kernel<<<blk_exact((long long)Etot * H1, 256), 256, 0, stream>>>(ei, E, Nn, As, Ad, Mk, H1);
  edge_expsum_kernel<<<blk_exact((long long)Etot * H1, 256), 256, 0, stream>>>(ei, E, Nn, As, Ad, Mk, Ss, H1);
  edge_agg_kernel<<<blk_exact(Etot, 8), 256, 0, stream>>>(ei, E, Nn, As, Ad, Mk, Ss, Hlin, Hout, H1, C1);
  bias_relu_kernel<<<blk_gs((size_t)Nn * Ct1, 256), 256, 0, stream>>>(Hout, b1, (size_t)Nn * Ct1, Ct1);

  // ===== GAT layer 2 =====
  f32_to_bf16_kernel<<<blk_gs((size_t)Nn * Ct1, 256), 256, 0, stream>>>(Hout, Abf, (size_t)Nn * Ct1);
  transpose_f32_to_bf16_kernel<<<blk_exact(Ct1 * HID, 256), 256, 0, stream>>>(w2, Wbf, Ct1, HID);
  gemm(Abf, Wbf, Hlin, nullptr, Nn, HID, Ct1, 0);                      // h2_lin
  alpha_kernel<<<blk_exact(Nn * H2, 256), 256, 0, stream>>>(Hlin, as2v, ad2v, As, Ad, Nn, H2, C2);
  fill_u32_kernel<<<blk_gs((size_t)Nn * H2, 256), 256, 0, stream>>>(Mk, 0u, (size_t)Nn * H2);
  fill_u32_kernel<<<blk_gs((size_t)Nn * H2, 256), 256, 0, stream>>>((unsigned*)Ss, 0u, (size_t)Nn * H2);
  fill_u32_kernel<<<blk_gs((size_t)Nn * Ct2, 256), 256, 0, stream>>>((unsigned*)Hout, 0u, (size_t)Nn * Ct2);
  edge_max_kernel<<<blk_exact((long long)Etot * H2, 256), 256, 0, stream>>>(ei, E, Nn, As, Ad, Mk, H2);
  edge_expsum_kernel<<<blk_exact((long long)Etot * H2, 256), 256, 0, stream>>>(ei, E, Nn, As, Ad, Mk, Ss, H2);
  edge_agg_kernel<<<blk_exact(Etot, 8), 256, 0, stream>>>(ei, E, Nn, As, Ad, Mk, Ss, Hlin, Hout, H2, C2);
  bias_relu_kernel<<<blk_gs((size_t)Nn * Ct2, 256), 256, 0, stream>>>(Hout, b2, (size_t)Nn * Ct2, Ct2);

  // ===== global mean pool =====
  fill_u32_kernel<<<blk_gs((size_t)G * HID, 256), 256, 0, stream>>>((unsigned*)Ps, 0u, (size_t)G * HID);
  fill_u32_kernel<<<1, 64, 0, stream>>>((unsigned*)Pc, 0u, (size_t)G);
  pool_sum_kernel<<<blk_exact((long long)Nn * HID, 256), 256, 0, stream>>>(Hout, batch, Ps, Pc, Nn, HID);
  pool_div_kernel<<<blk_exact(G * HID, 256), 256, 0, stream>>>(Ps, Pc, Gf, G, HID);

  // ===== vis = relu(visual @ wvp^T + bvp) =====  (wvp is [out,in] == Bt layout already)
  f32_to_bf16_kernel<<<blk_gs((size_t)G * VISD, 256), 256, 0, stream>>>(visual, Visb, (size_t)G * VISD);
  f32_to_bf16_kernel<<<blk_gs((size_t)HID * VISD, 256), 256, 0, stream>>>(wvp, Wbf, (size_t)HID * VISD);
  gemm(Visb, Wbf, Vis, bvp, G, HID, VISD, 1);

  // ===== MHA with S=1: attn_out = (Gf @ wv^T + bv) @ wo^T + bo =====
  f32_to_bf16_kernel<<<blk_gs((size_t)G * HID, 256), 256, 0, stream>>>(Gf, Gfb, (size_t)G * HID);
  f32_to_bf16_kernel<<<blk_gs((size_t)HID * HID, 256), 256, 0, stream>>>(wv, Wbf, (size_t)HID * HID);
  gemm(Gfb, Wbf, Vt, bv, G, HID, HID, 0);
  f32_to_bf16_kernel<<<blk_gs((size_t)G * HID, 256), 256, 0, stream>>>(Vt, Vtb, (size_t)G * HID);
  f32_to_bf16_kernel<<<blk_gs((size_t)HID * HID, 256), 256, 0, stream>>>(wo, Wbf, (size_t)HID * HID);
  gemm(Vtb, Wbf, At, bo, G, HID, HID, 0);

  // ===== fusion: out = relu([vis, attn_out] @ wf^T + bf) =====
  concat_bf16_kernel<<<blk_exact(G * 2 * HID, 256), 256, 0, stream>>>(Vis, At, Cmb, G, HID);
  f32_to_bf16_kernel<<<blk_gs((size_t)HID * 2 * HID, 256), 256, 0, stream>>>(wf, Wbf, (size_t)HID * 2 * HID);
  gemm(Cmb, Wbf, out, bff, G, HID, 2 * HID, 1);
}
